// ResTGCN_1855425872360
// MI455X (gfx1250) — matrix operand order repaired and checked by reading, not compiled, not run on hardware
//
#include <hip/hip_runtime.h>
#include <hip/hip_bf16.h>

// ---- CDNA5 WMMA vector types (probe-confirmed signatures) ------------------
typedef __attribute__((ext_vector_type(16))) __bf16 v16bf;
typedef __attribute__((ext_vector_type(8)))  float  v8f;

#define DIN  128
#define DOUT 64

__device__ __forceinline__ void wait_async0() {
#if __has_builtin(__builtin_amdgcn_s_wait_asynccnt)
  __builtin_amdgcn_s_wait_asynccnt(0);
#else
  asm volatile("s_wait_asynccnt 0x0" ::: "memory");
#endif
}

// ---------------------------------------------------------------------------
// Degree / dinv
// ---------------------------------------------------------------------------
__global__ void k_deg_init(float* __restrict__ deg, int n) {
  int i = blockIdx.x * blockDim.x + threadIdx.x;
  if (i < n) deg[i] = 1.0f;  // self-loop weight
}

__global__ void k_deg_acc(const int* __restrict__ dst, const float* __restrict__ ew,
                          float* __restrict__ deg, int ne) {
  int i = blockIdx.x * blockDim.x + threadIdx.x;
  if (i < ne) atomicAdd(&deg[dst[i]], ew[i]);
}

__global__ void k_deg_rsqrt(float* __restrict__ deg, int n) {
  int i = blockIdx.x * blockDim.x + threadIdx.x;
  if (i < n) deg[i] = rsqrtf(deg[i]);
}

// ---------------------------------------------------------------------------
// Pack a [128,64] f32 row-major weight into the 32-lane bf16 B-fragment layout
// for v_wmma_f32_16x16x32_bf16, split into hi/lo bf16 for fp32-class accuracy.
// One block (128 thr) per 16-col tile; layout per tile: [kchunk(4)][lane(32)][e(16)]
// B-frag mapping: lanes 0-15: N=lane,  K = kchunk*32 + e
//                 lanes 16-31: N=lane-16, K = kchunk*32 + 16 + e
// ---------------------------------------------------------------------------
__global__ void k_pack_b(const float* __restrict__ W,
                         __bf16* __restrict__ dstH, __bf16* __restrict__ dstL) {
  const int t    = blockIdx.x;
  const int tid  = threadIdx.x;      // 0..127
  const int c    = tid >> 5;         // k-chunk 0..3
  const int lane = tid & 31;
  const int colW  = (t << 4) + (lane & 15);
  const int kbase = (c << 5) + ((lane < 16) ? 0 : 16);
  const size_t off = (((size_t)t * 4 + c) * 32 + lane) * 16;
  __bf16* oh = dstH + off;
  __bf16* ol = dstL + off;
#pragma unroll
  for (int e = 0; e < 16; ++e) {
    float v  = W[(size_t)(kbase + e) * DOUT + colW];
    __bf16 h = (__bf16)v;
    oh[e] = h;
    ol[e] = (__bf16)(v - (float)h);
  }
}

// ---------------------------------------------------------------------------
// WMMA GEMM: Out[16 rows x ntiles*16 cols] = A[16 x 128] @ Bpack
// A row r: cols 0..63 from A0[r*lda0 + c], cols 64..127 from A1[r*lda1 + c].
// 128 threads = 4 wave32; block handles one 16-row M tile; waves split col tiles.
// A tile staged raw-f32 into LDS via GLOBAL_LOAD_ASYNC_TO_LDS_B128 (ASYNCcnt),
// bf16 hi/lo split done at fragment-build time in registers.
// fp32-via-bf16: acc += ah*bl + al*bh + ah*bh  (3 WMMAs / K-chunk).
// Requires nrows % 16 == 0 (N = 100000 = 6250 * 16).
// ---------------------------------------------------------------------------
__launch_bounds__(128)
__global__ void k_gemm_wmma(const float* __restrict__ A0, int lda0,
                            const float* __restrict__ A1, int lda1,
                            const __bf16* __restrict__ BpH,
                            const __bf16* __restrict__ BpL,
                            float* __restrict__ Out, int ldo, int ntiles) {
  __shared__ float sA[16 * 132];   // padded stride 132 spreads LDS banks

  const int tid     = threadIdx.x;
  const int rowbase = blockIdx.x << 4;

  { // async stage of the 16x128 f32 A tile: 4 x b128 per thread, no VGPR data
    const int r  = tid >> 3;               // 0..15
    const int c0 = (tid & 7) << 4;         // 0,16,...,112
    const float* src = (c0 < 64) ? (A0 + (size_t)(rowbase + r) * lda0 + c0)
                                 : (A1 + (size_t)(rowbase + r) * lda1 + (c0 - 64));
    unsigned ldsoff = (unsigned)(uintptr_t)(&sA[r * 132 + c0]);
    // INST_OFFSET adds to both the LDS and global address -> stride the copy
    asm volatile("global_load_async_to_lds_b128 %0, %1, off\n\t"
                 "global_load_async_to_lds_b128 %0, %1, off offset:16\n\t"
                 "global_load_async_to_lds_b128 %0, %1, off offset:32\n\t"
                 "global_load_async_to_lds_b128 %0, %1, off offset:48"
                 :: "v"(ldsoff), "v"(src) : "memory");
  }
  wait_async0();
  __syncthreads();

  const int lane  = tid & 31;
  const int wave  = tid >> 5;
  const int arow  = lane & 15;
  const int khalf = (lane < 16) ? 0 : 16;

  // A-frag mapping (16x32 bf16): lane<16: K=e
  //                              lane>=16: K=16+e
  v16bf ah[4], al[4];
#pragma unroll
  for (int c = 0; c < 4; ++c) {
    union { v16bf v; __bf16 e[16]; } uh, ul;
    const int base = arow * 132 + c * 32 + khalf;
#pragma unroll
    for (int e = 0; e < 8; ++e) {
      float f0 = sA[base + e];
      float f1 = sA[base + 8 + e];
      __bf16 h0 = (__bf16)f0;
      __bf16 h1 = (__bf16)f1;
      uh.e[e]     = h0;
      uh.e[e + 8] = h1;
      ul.e[e]     = (__bf16)(f0 - (float)h0);
      ul.e[e + 8] = (__bf16)(f1 - (float)h1);
    }
    ah[c] = uh.v;
    al[c] = ul.v;
  }

  for (int t = wave; t < ntiles; t += 4) {
    v8f acc = {};
    const size_t tbase = (((size_t)t * 4) * 32 + lane) * 16;
#pragma unroll
    for (int c = 0; c < 4; ++c) {
      v16bf bh = *(const v16bf*)(BpH + tbase + (size_t)c * 512);
      v16bf bl = *(const v16bf*)(BpL + tbase + (size_t)c * 512);
      acc = __builtin_amdgcn_wmma_f32_16x16x32_bf16(false, ah[c], false, bl,
                                                    (short)0, acc, false, false);
      acc = __builtin_amdgcn_wmma_f32_16x16x32_bf16(false, al[c], false, bh,
                                                    (short)0, acc, false, false);
      acc = __builtin_amdgcn_wmma_f32_16x16x32_bf16(false, ah[c], false, bh,
                                                    (short)0, acc, false, false);
    }
    // C/D layout: VGPR j -> rows (j | j+8); lanes 0-15 cols, 16-31 rows+8
    const int col = (t << 4) + (lane & 15);
    const int r0  = rowbase + ((lane < 16) ? 0 : 8);
    float* o = Out + (size_t)r0 * ldo + col;
#pragma unroll
    for (int j = 0; j < 8; ++j) o[(size_t)j * ldo] = acc[j];
  }
}

// ---------------------------------------------------------------------------
// AGG init: self-loop + conv bias  (avoids separate zero pass before scatter)
// ---------------------------------------------------------------------------
__global__ void k_agg_init(const float* __restrict__ XW, const float* __restrict__ dinv,
                           const float* __restrict__ bcz, const float* __restrict__ bcr,
                           const float* __restrict__ bch, float* __restrict__ AGG,
                           int total /* n*192 */) {
  int idx = blockIdx.x * blockDim.x + threadIdx.x;
  if (idx >= total) return;
  int node = idx / 192;
  int c    = idx - node * 192;
  int g = c >> 6, cc = c & 63;
  float b  = (g == 0) ? bcz[cc] : ((g == 1) ? bcr[cc] : bch[cc]);
  float di = dinv[node];
  AGG[idx] = di * di * XW[idx] + b;
}

// ---------------------------------------------------------------------------
// Edge scatter: one wave32 per edge, 6 coalesced atomic rounds of 32 floats.
// Roofline bottleneck (307M f32 atomics + random 768B gathers; XW slab is
// mostly L2-resident) -> prefetch next edge's source row to deepen the pipe.
// ---------------------------------------------------------------------------
__global__ void k_scatter(const int* __restrict__ src, const int* __restrict__ dst,
                          const float* __restrict__ ew, const float* __restrict__ dinv,
                          const float* __restrict__ XW, float* __restrict__ AGG, int ne) {
  int gid  = blockIdx.x * blockDim.x + threadIdx.x;
  int lane = gid & 31;
  int wid  = gid >> 5;
  int nw   = (gridDim.x * blockDim.x) >> 5;
  for (int e = wid; e < ne; e += nw) {
    int e2 = e + nw;
    if (e2 < ne) {  // global_prefetch of next iteration's gather row
      int s2 = src[e2];
      __builtin_prefetch(XW + (size_t)s2 * 192 + lane, 0, 1);
    }
    int s = src[e], d = dst[e];
    float norm = dinv[s] * ew[e] * dinv[d];
    const float* xs = XW  + (size_t)s * 192;
    float*       ad = AGG + (size_t)d * 192;
#pragma unroll
    for (int i = 0; i < 6; ++i) {
      int c = lane + i * 32;
      atomicAdd(ad + c, norm * xs[c]);
    }
  }
}

// ---------------------------------------------------------------------------
// Z / R gates.  Zb aliases Cr (read-before-write, same thread => safe).
// ---------------------------------------------------------------------------
__global__ void k_zr(const float* Cz, const float* Cr,
                     const float* __restrict__ AGG, const float* __restrict__ H,
                     const float* __restrict__ blz, const float* __restrict__ blr,
                     float* Zb, float* __restrict__ RH, int total /* n*64 */) {
  int idx = blockIdx.x * blockDim.x + threadIdx.x;
  if (idx >= total) return;
  int node = idx >> 6, c = idx & 63;
  float xz = AGG[(size_t)node * 192 + c];
  float xr = AGG[(size_t)node * 192 + 64 + c];
  float cz = Cz[idx], cr = Cr[idx];
  float z = 1.0f / (1.0f + __expf(-(cz + blz[c] + xz)));
  float r = 1.0f / (1.0f + __expf(-(cr + blr[c] + xr)));
  Zb[idx] = z;
  RH[idx] = r * H[idx];
}

// ---------------------------------------------------------------------------
// Final blend: out = Z*H + (1-Z)*(tanh(Ch + bl_h) + H_last)
// ---------------------------------------------------------------------------
__global__ void k_final(const float* __restrict__ Ch, const float* __restrict__ AGG,
                        const float* __restrict__ Zb, const float* __restrict__ H,
                        const float* __restrict__ blh, float* __restrict__ out,
                        int total /* n*64 */) {
  int idx = blockIdx.x * blockDim.x + threadIdx.x;
  if (idx >= total) return;
  int node = idx >> 6, c = idx & 63;
  float hl = AGG[(size_t)node * 192 + 128 + c];
  float ht = tanhf(Ch[idx] + blh[c]) + hl;
  float z  = Zb[idx];
  out[idx] = z * H[idx] + (1.0f - z) * ht;
}

// ---------------------------------------------------------------------------
extern "C" void kernel_launch(void* const* d_in, const int* in_sizes, int n_in,
                              void* d_out, int out_size, void* d_ws, size_t ws_size,
                              hipStream_t stream) {
  const float* X    = (const float*)d_in[0];
  const int*   ei   = (const int*)d_in[1];     // [2,E] (src row, dst row)
  const float* ew   = (const float*)d_in[2];
  const float* H    = (const float*)d_in[3];
  const float* Wc_z = (const float*)d_in[4];
  const float* bc_z = (const float*)d_in[5];
  const float* Wl_z = (const float*)d_in[6];
  const float* bl_z = (const float*)d_in[7];
  const float* Wc_r = (const float*)d_in[8];
  const float* bc_r = (const float*)d_in[9];
  const float* Wl_r = (const float*)d_in[10];
  const float* bl_r = (const float*)d_in[11];
  const float* Wc_h = (const float*)d_in[12];
  const float* bc_h = (const float*)d_in[13];
  const float* Wl_h = (const float*)d_in[14];
  const float* bl_h = (const float*)d_in[15];

  const int n  = in_sizes[0] / DIN;   // 100000 (multiple of 16)
  const int ne = in_sizes[2];         // 1600000
  const int* src = ei;
  const int* dst = ei + ne;

  // ---- workspace layout (floats), ~155 MB total -------------------------
  float* ws = (float*)d_ws;
  size_t o = 0;
  float* dinv = ws + o; o += ((size_t)n + 63) & ~(size_t)63;
  float* AGG  = ws + o; o += (size_t)n * 192;
  float* XW   = ws + o; o += (size_t)n * 192;
  // XW is consumed by agg_init + scatter; reuse its slab afterwards:
  float* Cz = XW;                         // z-linear out, later reused for Ch
  float* Cr = XW + (size_t)n * 64;        // r-linear out, later reused for Z
  float* RH = XW + (size_t)2 * n * 64;    // R * H
  float* Zb = Cr;
  float* Ch = Cz;
  __bf16* packs = (__bf16*)(ws + o);
  const size_t TILE = 4 * 32 * 16;        // bf16 elems per 16-col tile (2048)
  __bf16* BcH  = packs;                   // 12 tiles (z|r|h conv weights)
  __bf16* BcL  = BcH  + 12 * TILE;
  __bf16* BlzH = BcL  + 12 * TILE;  __bf16* BlzL = BlzH + 4 * TILE;
  __bf16* BlrH = BlzL + 4 * TILE;   __bf16* BlrL = BlrH + 4 * TILE;
  __bf16* BlhH = BlrL + 4 * TILE;   __bf16* BlhL = BlhH + 4 * TILE;

  const int T256n  = (n + 255) / 256;
  const int T256e  = (ne + 255) / 256;
  const int nt192  = n * 192;
  const int nt64   = n * 64;

  // 1. degree + dinv
  k_deg_init <<<T256n, 256, 0, stream>>>(dinv, n);
  k_deg_acc  <<<T256e, 256, 0, stream>>>(dst, ew, dinv, ne);
  k_deg_rsqrt<<<T256n, 256, 0, stream>>>(dinv, n);

  // 2. pre-swizzle weights into WMMA B-fragment layout (hi/lo bf16)
  k_pack_b<<<4, 128, 0, stream>>>(Wc_z, BcH,            BcL);
  k_pack_b<<<4, 128, 0, stream>>>(Wc_r, BcH + 4 * TILE, BcL + 4 * TILE);
  k_pack_b<<<4, 128, 0, stream>>>(Wc_h, BcH + 8 * TILE, BcL + 8 * TILE);
  k_pack_b<<<4, 128, 0, stream>>>(Wl_z, BlzH, BlzL);
  k_pack_b<<<4, 128, 0, stream>>>(Wl_r, BlrH, BlrL);
  k_pack_b<<<4, 128, 0, stream>>>(Wl_h, BlhH, BlhL);

  // 3. XW = X @ [Wc_z | Wc_r | Wc_h]   ([N,128] @ [128,192])
  k_gemm_wmma<<<n / 16, 128, 0, stream>>>(X, DIN, X + 64, DIN,
                                          BcH, BcL, XW, 192, 12);

  // 4. AGG = dinv^2 * XW + bias (self loop), then edge scatter
  k_agg_init<<<(nt192 + 255) / 256, 256, 0, stream>>>(XW, dinv, bc_z, bc_r, bc_h,
                                                      AGG, nt192);
  k_scatter<<<2048, 256, 0, stream>>>(src, dst, ew, dinv, XW, AGG, ne);

  // 5. gate linears z, r:  [AGGg | H] @ Wl_g
  k_gemm_wmma<<<n / 16, 128, 0, stream>>>(AGG,      192, H, DOUT,
                                          BlzH, BlzL, Cz, DOUT, 4);
  k_gemm_wmma<<<n / 16, 128, 0, stream>>>(AGG + 64, 192, H, DOUT,
                                          BlrH, BlrL, Cr, DOUT, 4);
  k_zr<<<(nt64 + 255) / 256, 256, 0, stream>>>(Cz, Cr, AGG, H, bl_z, bl_r,
                                               Zb, RH, nt64);

  // 6. candidate:  [H_last | R*H] @ Wl_h, then final blend
  k_gemm_wmma<<<n / 16, 128, 0, stream>>>(AGG + 128, 192, RH, DOUT,
                                          BlhH, BlhL, Ch, DOUT, 4);
  k_final<<<(nt64 + 255) / 256, 256, 0, stream>>>(Ch, AGG, Zb, H, bl_h,
                                                  (float*)d_out, nt64);
}
